// FasterRCNN_53489522704686
// MI455X (gfx1250) — compile-verified, matrix operand order checked by reading, not run on hardware
//
#include <hip/hip_runtime.h>
#include <hip/hip_bf16.h>
#include <math.h>

typedef __bf16 bf16;
typedef __attribute__((ext_vector_type(8)))  bf16  v8bf;
typedef __attribute__((ext_vector_type(16))) bf16  v16bf;
typedef __attribute__((ext_vector_type(8)))  float v8f;

#define BATCH 4
#define HF    128
#define WF    128
#define NPIX  (BATCH*HF*WF)        // 65536 feature pixels
#define NANCH 9
#define NIMG  (HF*WF*NANCH)        // 147456 anchors per image
#define TOPK  2000
#define IMG   512

// ---------------------------------------------------------------------------
// conv1: 7x7 stride2 pad3, 3->64, relu.  NCHW f32 in, NHWC bf16 out [4,256,256,64]
// ---------------------------------------------------------------------------
__global__ void conv1_kernel(const float* __restrict__ img,
                             const float* __restrict__ w1,
                             const float* __restrict__ b1,
                             bf16* __restrict__ out) {
  size_t t = (size_t)blockIdx.x * 256 + threadIdx.x;
  if (t >= (size_t)BATCH * 256 * 256 * 64) return;
  int co = t & 63;
  size_t p = t >> 6;
  int x = p & 255; p >>= 8;
  int y = p & 255;
  int b = (int)(p >> 8);
  float acc = b1[co];
  int y0 = y * 2 - 3, x0 = x * 2 - 3;
  for (int ci = 0; ci < 3; ++ci) {
    for (int ky = 0; ky < 7; ++ky) {
      int yy = y0 + ky;
      if ((unsigned)yy >= 512u) continue;
      for (int kx = 0; kx < 7; ++kx) {
        int xx = x0 + kx;
        if ((unsigned)xx >= 512u) continue;
        acc += img[(((size_t)b * 3 + ci) * 512 + yy) * 512 + xx] *
               w1[((co * 3 + ci) * 7 + ky) * 7 + kx];
      }
    }
  }
  out[t] = (bf16)(acc > 0.f ? acc : 0.f);
}

// ---------------------------------------------------------------------------
// maxpool 3x3 stride2 pad1 on NHWC bf16: [4,256,256,64] -> [4,128,128,64]
// ---------------------------------------------------------------------------
__global__ void maxpool_kernel(const bf16* __restrict__ in, bf16* __restrict__ out) {
  size_t t = (size_t)blockIdx.x * 256 + threadIdx.x;
  if (t >= (size_t)BATCH * 128 * 128 * 64) return;
  int c = t & 63;
  size_t p = t >> 6;
  int x = p & 127; p >>= 7;
  int y = p & 127;
  int b = (int)(p >> 7);
  float m = -3.0e38f;
  for (int ky = 0; ky < 3; ++ky) {
    int yy = 2 * y - 1 + ky;
    if ((unsigned)yy >= 256u) continue;
    for (int kx = 0; kx < 3; ++kx) {
      int xx = 2 * x - 1 + kx;
      if ((unsigned)xx >= 256u) continue;
      float v = (float)in[(((size_t)b * 256 + yy) * 256 + xx) * 64 + c];
      m = v > m ? v : m;
    }
  }
  out[t] = (bf16)m;
}

// ---------------------------------------------------------------------------
// Weight fragment prep: f32 OIHW -> bf16 WMMA B-fragment layout.
// Layout: frag[nt][tap][kb][lane][16], one 32B vector per lane per fragment.
// 16-bit B striping: lane gives column n=lane&15, half=lane>>4 selects K runs:
//   e<8  -> k = half*8 + e
//   e>=8 -> k = 16 + half*8 + (e-8)
// ---------------------------------------------------------------------------
__global__ void wprep_kernel(const float* __restrict__ W, bf16* __restrict__ F,
                             int Cin, int Cout) {
  int KB = Cin >> 5, NT = Cout >> 4;
  int total = NT * 9 * KB * 32;
  int t = blockIdx.x * 256 + threadIdx.x;
  if (t >= total) return;
  int lane = t & 31;
  int r = t >> 5;
  int kb = r % KB; r /= KB;
  int tap = r % 9;
  int nt = r / 9;
  int n = lane & 15, half = lane >> 4;
  int cout = nt * 16 + n;
  bf16* dst = F + (size_t)t * 16;
#pragma unroll
  for (int e = 0; e < 16; ++e) {
    int k = (e < 8) ? (half * 8 + e) : (16 + half * 8 + (e - 8));
    int cin = kb * 32 + k;
    dst[e] = (bf16)W[((size_t)cout * Cin + cin) * 9 + tap];
  }
}

// ---------------------------------------------------------------------------
// Implicit-GEMM 3x3 pad1 conv + relu on NHWC bf16, wave32 WMMA.
// Workgroup = 8 waves = one full 128-pixel row (fixed b,y), 64 couts (4 n-tiles).
// K-loop kb-outer / tap-inner; per-kb 36KB weight block double-buffered in LDS
// via global_load_async_to_lds_b128 (ASYNCcnt), consumed via ds_load.
// A fragment (16x32 bf16): lanes 0-15 rows, per-lane two contiguous 8-channel
// runs at channel offsets half*8 and 16+half*8 within the 32-wide K block.
// ---------------------------------------------------------------------------
#define WBLK 36864  // 4 ntiles * 9 taps * 32 lanes * 32B

template <int CIN, int COUT>
__global__ void __launch_bounds__(256) conv3x3_wmma(
    const bf16* __restrict__ in, const bf16* __restrict__ wf,
    const float* __restrict__ bias, bf16* __restrict__ out) {
  constexpr int KB = CIN / 32;
  extern __shared__ char smem[];
  const unsigned smbase = (unsigned)(uintptr_t)smem;
  const int tid = threadIdx.x;
  const int lane = tid & 31;
  const int wid = tid >> 5;
  const int y = blockIdx.x & 127;
  const int b = blockIdx.x >> 7;
  const int ng = blockIdx.y;          // group of 4 consecutive n-tiles
  const int x0 = wid << 4;
  const int half = lane >> 4;
  const int mrow = lane & 15;
  const int ncol = lane & 15;

  // async-stage one kb weight block (36KB) into LDS buffer buf.
  // region layout: [(j*9+tap)][lane][32B], chunk stride 1KB; 2304 x 16B xfers.
  auto stage = [&](int kb, int buf) {
#pragma unroll
    for (int rd = 0; rd < 9; ++rd) {
      int id = rd * 256 + tid;        // 0..2303
      int chunk = id >> 6;            // j*9+tap
      int o = (id & 63) << 4;         // byte offset within chunk
      const char* g = (const char*)wf +
          (((size_t)(ng * 36 + chunk)) * KB + kb) * 1024 + o;
      unsigned l = smbase + buf * WBLK + chunk * 1024 + o;
      asm volatile("global_load_async_to_lds_b128 %0, %1, off"
                   :: "v"(l), "v"((unsigned long long)(uintptr_t)g)
                   : "memory");
    }
  };

  v8bf zero8;
#pragma unroll
  for (int e = 0; e < 8; ++e) zero8[e] = (bf16)0.0f;

  v8f acc[4];
#pragma unroll
  for (int j = 0; j < 4; ++j) {
    float bv = bias[(ng * 4 + j) * 16 + ncol];
#pragma unroll
    for (int r = 0; r < 8; ++r) acc[j][r] = bv;
  }

  stage(0, 0);
  for (int kb = 0; kb < KB; ++kb) {
    const int buf = kb & 1;
    if (kb + 1 < KB) {
      stage(kb + 1, buf ^ 1);
      // 9 outstanding = the batch just issued; previous batch has landed
      asm volatile("s_wait_asynccnt 9" ::: "memory");
    } else {
      asm volatile("s_wait_asynccnt 0" ::: "memory");
    }
    __syncthreads();
#pragma unroll
    for (int tap = 0; tap < 9; ++tap) {
      const int ky = tap / 3, kx = tap % 3;
      const int yy = y + ky - 1;
      const int xx = x0 + mrow + kx - 1;
      const bool inb = ((unsigned)yy < (unsigned)HF) && ((unsigned)xx < (unsigned)WF);
      const bf16* ap = in + ((long)(b * HF + yy) * WF + xx) * CIN + kb * 32 + half * 8;
      union { v16bf v; v8bf h[2]; } a;
      if (inb) {
        a.h[0] = *(const v8bf*)(ap);
        a.h[1] = *(const v8bf*)(ap + 16);
      } else {
        a.h[0] = zero8;
        a.h[1] = zero8;
      }
#pragma unroll
      for (int j = 0; j < 4; ++j) {
        const v16bf bm = *(const v16bf*)(smem + buf * WBLK + (j * 9 + tap) * 1024 + lane * 32);
        acc[j] = __builtin_amdgcn_wmma_f32_16x16x32_bf16(
            false, a.v, false, bm, (short)0, acc[j], false, false);
      }
    }
    __syncthreads();
  }

  // C/D layout: vgpr r, lane l -> m = r + 8*(l>>4), n = l&15
  const size_t rowbase = (size_t)(b * HF + y) * WF;
#pragma unroll
  for (int r = 0; r < 8; ++r) {
    const int x = x0 + r + half * 8;
    bf16* op = out + (rowbase + x) * COUT + ncol;
#pragma unroll
    for (int j = 0; j < 4; ++j) {
      float v = acc[j][r];
      v = v > 0.f ? v : 0.f;
      op[(ng * 4 + j) * 16] = (bf16)v;
    }
  }
}

// ---------------------------------------------------------------------------
// RPN head: 1x1 convs, 512 -> 9 (cls logits, f32) and 512 -> 36 (deltas, f32)
// ---------------------------------------------------------------------------
__global__ void head_kernel(const bf16* __restrict__ h,
                            const float* __restrict__ cw, const float* __restrict__ cb,
                            const float* __restrict__ bw, const float* __restrict__ bb,
                            float* __restrict__ obj, float* __restrict__ dlt) {
  size_t t = (size_t)blockIdx.x * 256 + threadIdx.x;
  if (t >= (size_t)NPIX * 45) return;
  int oc = (int)(t % 45);
  size_t pix = t / 45;
  const bf16* hp = h + pix * 512;
  const float* w = (oc < 9) ? (cw + (size_t)oc * 512) : (bw + (size_t)(oc - 9) * 512);
  float acc = (oc < 9) ? cb[oc] : bb[oc - 9];
  for (int c = 0; c < 512; ++c) acc += (float)hp[c] * w[c];
  if (oc < 9) obj[pix * 9 + oc] = acc;
  else        dlt[pix * 36 + (oc - 9)] = acc;
}

// ---------------------------------------------------------------------------
// top-k via score histogram (sigmoid monotonic; 1024 bins of s in (0,1))
// ---------------------------------------------------------------------------
__global__ void zero_kernel(int* p, int n) {
  int t = blockIdx.x * 256 + threadIdx.x;
  if (t < n) p[t] = 0;
}

__device__ __forceinline__ int score_bin(float logit) {
  float s = 1.f / (1.f + __expf(-logit));
  int bin = (int)(s * 1024.f);
  bin = bin < 0 ? 0 : (bin > 1023 ? 1023 : bin);
  return bin;
}

__global__ void hist_kernel(const float* __restrict__ obj, int* __restrict__ hist) {
  size_t t = (size_t)blockIdx.x * 256 + threadIdx.x;
  if (t >= (size_t)BATCH * NIMG) return;
  int b = (int)(t / NIMG);
  atomicAdd(&hist[b * 1024 + score_bin(obj[t])], 1);
}

__global__ void thresh_kernel(const int* __restrict__ hist, int* thr, int* above) {
  int b = threadIdx.x;
  if (b >= BATCH) return;
  const int* h = hist + b * 1024;
  int cum = 0, bin = 0, ab = 0;
  for (int i = 1023; i >= 0; --i) {
    int c = h[i];
    if (cum + c >= TOPK) { bin = i; ab = cum; break; }
    cum += c;
  }
  thr[b] = bin;
  above[b] = ab;
}

__global__ void select_kernel(const float* __restrict__ obj,
                              const int* __restrict__ thr, const int* __restrict__ above,
                              int* __restrict__ cnts, int* __restrict__ tidx) {
  size_t t = (size_t)blockIdx.x * 256 + threadIdx.x;
  if (t >= (size_t)BATCH * NIMG) return;
  int b = (int)(t / NIMG);
  int n = (int)(t % NIMG);
  int bin = score_bin(obj[t]);
  int tb = thr[b], ab = above[b];
  if (bin > tb) {
    int slot = atomicAdd(&cnts[b * 2], 1);
    if (slot < TOPK) tidx[b * TOPK + slot] = n;
  } else if (bin == tb) {
    int slot = atomicAdd(&cnts[b * 2 + 1], 1);
    int pos = ab + slot;
    if (pos < TOPK) tidx[b * TOPK + pos] = n;
  }
}

// ---------------------------------------------------------------------------
// decode proposals: anchors computed analytically, deltas applied
// ---------------------------------------------------------------------------
__global__ void proposals_kernel(const int* __restrict__ tidx,
                                 const float* __restrict__ dlt,
                                 float* __restrict__ out) {
  int t = blockIdx.x * 256 + threadIdx.x;
  if (t >= BATCH * TOPK) return;
  int b = t / TOPK;
  int idx = tidx[t];
  int a = idx % 9;
  int pix = idx / 9;
  int x = pix % WF;
  int y = pix / WF;
  const float sc[3] = {8.f, 16.f, 32.f};
  const float rt[3] = {0.5f, 1.f, 2.f};
  float s = sc[a / 3], r = rt[a % 3];
  float bw = sqrtf(s * s / r);
  float bh = bw * r;
  float cx = x * 16.f + 8.f, cy = y * 16.f + 8.f;
  float ax0 = cx - 0.5f * bw, ay0 = cy - 0.5f * bh;
  float ax1 = cx + 0.5f * bw, ay1 = cy + 0.5f * bh;
  ax0 = fminf(fmaxf(ax0, 0.f), (float)IMG);
  ay0 = fminf(fmaxf(ay0, 0.f), (float)IMG);
  ax1 = fminf(fmaxf(ax1, 0.f), (float)IMG);
  ay1 = fminf(fmaxf(ay1, 0.f), (float)IMG);
  float aw = ax1 - ax0, ah = ay1 - ay0;
  float axc = ax0 + 0.5f * aw, ayc = ay0 + 0.5f * ah;
  const float* d = dlt + ((size_t)b * (HF * WF) + pix) * 36 + a * 4;
  float px = d[0] * aw + axc;
  float py = d[1] * ah + ayc;
  float pw = __expf(d[2]) * aw;
  float ph = __expf(d[3]) * ah;
  float* o = out + (size_t)t * 4;
  o[0] = px - 0.5f * pw;
  o[1] = py - 0.5f * ph;
  o[2] = px + 0.5f * pw;
  o[3] = py + 0.5f * ph;
}

// ---------------------------------------------------------------------------
extern "C" void kernel_launch(void* const* d_in, const int* in_sizes, int n_in,
                              void* d_out, int out_size, void* d_ws, size_t ws_size,
                              hipStream_t stream) {
  (void)in_sizes; (void)n_in; (void)out_size; (void)ws_size;
  const float* images = (const float*)d_in[0];
  const float* w1 = (const float*)d_in[1];
  const float* b1 = (const float*)d_in[2];
  const float* w2 = (const float*)d_in[3];
  const float* b2 = (const float*)d_in[4];
  const float* w3 = (const float*)d_in[5];
  const float* b3 = (const float*)d_in[6];
  const float* w4 = (const float*)d_in[7];
  const float* b4 = (const float*)d_in[8];
  const float* rpn_w = (const float*)d_in[9];
  const float* rpn_b = (const float*)d_in[10];
  const float* cls_w = (const float*)d_in[11];
  const float* cls_b = (const float*)d_in[12];
  const float* box_w = (const float*)d_in[13];
  const float* box_b = (const float*)d_in[14];
  float* outp = (float*)d_out;

  char* ws = (char*)d_ws;
  size_t off = 0;
  auto alloc = [&](size_t bytes) -> char* {
    char* p = ws + off;
    off = (off + bytes + 255) & ~(size_t)255;
    return p;
  };
  bf16* x1 = (bf16*)alloc((size_t)BATCH * 256 * 256 * 64 * 2);  // conv1 out
  bf16* p1 = (bf16*)alloc((size_t)BATCH * 128 * 128 * 64 * 2);  // pool out
  bf16* x2 = (bf16*)alloc((size_t)BATCH * 128 * 128 * 128 * 2);
  bf16* x3 = (bf16*)alloc((size_t)BATCH * 128 * 128 * 256 * 2);
  bf16* x4 = (bf16*)alloc((size_t)BATCH * 128 * 128 * 512 * 2); // feats
  bf16* hb = (bf16*)alloc((size_t)BATCH * 128 * 128 * 512 * 2); // rpn hidden
  float* obj = (float*)alloc((size_t)BATCH * NIMG * 4);
  float* dlt = (float*)alloc((size_t)BATCH * NIMG * 4 * 4);
  bf16* wf2 = (bf16*)alloc((size_t)128 * 9 * 64 * 2);
  bf16* wf3 = (bf16*)alloc((size_t)256 * 9 * 128 * 2);
  bf16* wf4 = (bf16*)alloc((size_t)512 * 9 * 256 * 2);
  bf16* wfr = (bf16*)alloc((size_t)512 * 9 * 512 * 2);
  int* ctl = (int*)alloc((size_t)(4096 + 8 + 4 + 4 + BATCH * TOPK) * 4);
  int* hist = ctl;
  int* cnts = hist + 4096;
  int* thr = cnts + 8;
  int* above = thr + 4;
  int* tidx = above + 4;
  const int nctl = 4096 + 8 + 4 + 4 + BATCH * TOPK;

  // zero control block
  zero_kernel<<<(nctl + 255) / 256, 256, 0, stream>>>(ctl, nctl);

  // weight fragment prep
  wprep_kernel<<<(8 * 9 * 2 * 32 + 255) / 256, 256, 0, stream>>>(w2, wf2, 64, 128);
  wprep_kernel<<<(16 * 9 * 4 * 32 + 255) / 256, 256, 0, stream>>>(w3, wf3, 128, 256);
  wprep_kernel<<<(32 * 9 * 8 * 32 + 255) / 256, 256, 0, stream>>>(w4, wf4, 256, 512);
  wprep_kernel<<<(32 * 9 * 16 * 32 + 255) / 256, 256, 0, stream>>>(rpn_w, wfr, 512, 512);

  // backbone
  conv1_kernel<<<(BATCH * 256 * 256 * 64) / 256, 256, 0, stream>>>(images, w1, b1, x1);
  maxpool_kernel<<<(BATCH * 128 * 128 * 64) / 256, 256, 0, stream>>>(x1, p1);

  const size_t shmem = 2 * WBLK;  // 72KB double-buffered weight block
  dim3 g2(BATCH * HF, 128 / 64);
  conv3x3_wmma<64, 128><<<g2, 256, shmem, stream>>>(p1, wf2, b2, x2);
  dim3 g3(BATCH * HF, 256 / 64);
  conv3x3_wmma<128, 256><<<g3, 256, shmem, stream>>>(x2, wf3, b3, x3);
  dim3 g4(BATCH * HF, 512 / 64);
  conv3x3_wmma<256, 512><<<g4, 256, shmem, stream>>>(x3, wf4, b4, x4);
  dim3 gr(BATCH * HF, 512 / 64);
  conv3x3_wmma<512, 512><<<gr, 256, shmem, stream>>>(x4, wfr, rpn_b, hb);

  // RPN head
  head_kernel<<<((size_t)NPIX * 45 + 255) / 256, 256, 0, stream>>>(
      hb, cls_w, cls_b, box_w, box_b, obj, dlt);

  // top-k selection
  hist_kernel<<<((size_t)BATCH * NIMG + 255) / 256, 256, 0, stream>>>(obj, hist);
  thresh_kernel<<<1, BATCH, 0, stream>>>(hist, thr, above);
  select_kernel<<<((size_t)BATCH * NIMG + 255) / 256, 256, 0, stream>>>(
      obj, thr, above, cnts, tidx);

  // decode
  proposals_kernel<<<(BATCH * TOPK + 255) / 256, 256, 0, stream>>>(tidx, dlt, outp);
}
